// AttnSAN_89361089561153
// MI455X (gfx1250) — matrix operand order repaired and checked by reading, not compiled, not run on hardware
//
#include <hip/hip_runtime.h>
#include <hip/hip_fp16.h>

typedef __attribute__((ext_vector_type(16))) _Float16 v16h;
typedef __attribute__((ext_vector_type(8)))  _Float16 v8h;
typedef __attribute__((ext_vector_type(8)))  float    v8f;

#define NB 32
#define NQ 48
#define NKC 36
#define ND 1024

__device__ __forceinline__ v16h hcat(v8h lo, v8h hi) {
    v16h r;
#pragma unroll
    for (int i = 0; i < 8; ++i) { r[i] = lo[i]; r[i + 8] = hi[i]; }
    return r;
}

// Branch-free fast math on CDNA5 trans units
__device__ __forceinline__ float fast_sigmoid(float x) {
    return __builtin_amdgcn_rcpf(1.0f + __builtin_amdgcn_exp2f(-1.442695041f * x));
}
__device__ __forceinline__ float fast_tanh(float x) {
#if __has_builtin(__builtin_amdgcn_tanhf)
    return __builtin_amdgcn_tanhf(x);          // native v_tanh_f32 on gfx1250
#else
    float t = __builtin_amdgcn_exp2f(2.885390082f * x);
    return 1.0f - 2.0f * __builtin_amdgcn_rcpf(1.0f + t);
#endif
}

// reuse hint must be an immediate -> template parameter
template <bool RB>
__device__ __forceinline__ v8f wmma_f16(v16h a, v16h b, v8f c) {
    return __builtin_amdgcn_wmma_f32_16x16x32_f16(false, a, false, b, (short)0, c,
                                                  false, RB);
}

// ---------------------------------------------------------------------------
// Generic fp32 -> f16 conversion (n divisible by 4)
// ---------------------------------------------------------------------------
__global__ void convert_f16(const float* __restrict__ src, _Float16* __restrict__ dst, int n) {
    int i = (blockIdx.x * blockDim.x + threadIdx.x) * 4;
    if (i + 3 < n) {
        float4 v = *(const float4*)(src + i);
        dst[i + 0] = (_Float16)v.x; dst[i + 1] = (_Float16)v.y;
        dst[i + 2] = (_Float16)v.z; dst[i + 3] = (_Float16)v.w;
    }
}

// ---------------------------------------------------------------------------
// Projection GEMM (all-f16 operands): out[m,e] = Xh[m,:] . Wh[e,:] + bias[e]
// mode 1: out = tanh(out)*wcom[e].  One 16x16 tile per wave.
// grid = (M/16, 64/4), block = 128 (4 waves).
// ---------------------------------------------------------------------------
__global__ void proj_gemm(const _Float16* __restrict__ Xh, const _Float16* __restrict__ Wh,
                          const float* __restrict__ bias, const float* __restrict__ wcom,
                          float* __restrict__ out, int mode) {
    const int wave = threadIdx.x >> 5;
    const int lane = threadIdx.x & 31;
    const int low  = lane & 15;
    const int hi   = lane >> 4;

    const int mtile = blockIdx.x;
    const int ntile = blockIdx.y * 4 + wave;
    const int e     = ntile * 16 + low;       // output column this lane owns (B frag)
    const int arow  = mtile * 16 + low;       // A row this lane owns

    v8f acc = {};
    const _Float16* ap = Xh + (size_t)arow * ND;
    const _Float16* bp = Wh + (size_t)e * ND;

#pragma unroll 2
    for (int kk = 0; kk < 32; ++kk) {
        const int k0 = kk * 32;
        // A frag: chunks at k0 + hi*8 and +16 halves (ISA 16-bit A layout)
        const _Float16* a0p = ap + k0 + hi * 8;
        v16h af = hcat(*(const v8h*)(a0p), *(const v8h*)(a0p + 16));
        // B frag: chunks at k0 + hi*8 and +16 halves (matches A layout)
        const _Float16* b0p = bp + k0 + hi * 8;
        v16h bf = hcat(*(const v8h*)(b0p), *(const v8h*)(b0p + 16));
        acc = wmma_f16<false>(af, bf, acc);
    }

    const float bv = bias[e];
    const float wv = (mode == 1) ? wcom[e] : 0.0f;
#pragma unroll
    for (int r = 0; r < 8; ++r) {
        int row = mtile * 16 + r + hi * 8;
        float v = acc[r] + bv;
        if (mode == 1) v = fast_tanh(v) * wv;
        out[(size_t)row * ND + e] = v;
    }
}

// ---------------------------------------------------------------------------
// Fused main GEMM: one WG per (b, k-pair). Two sigmoid A-matrices staged to LDS
// (2 x 48x1024 f16 = 192KB; CDNA5 allows 320KB/WG). 512 threads = 16 waves:
// waves 0-7 work on k0, waves 8-15 on k1, walking identical W0 rows in lockstep
// so the second group hits hot cache lines (halves W0 L2 traffic vs 1-k WGs).
// scores[b,q,k] = sum_e tanh(A@W0^T + b0)*g[b,k,e].
// ---------------------------------------------------------------------------
__global__ void fused_score_gemm(const float* __restrict__ qproj, const float* __restrict__ cproj,
                                 const float* __restrict__ g, const _Float16* __restrict__ w0h,
                                 const float* __restrict__ b0, float* __restrict__ scores) {
    __shared__ _Float16 As[2][NQ * ND];   // 192 KB

    const int b     = blockIdx.x / (NKC / 2);
    const int kpair = blockIdx.x % (NKC / 2);
    const int tid   = threadIdx.x;

    // Stage both A matrices; qproj row loaded once, reused for both k's
    const float* qrow  = qproj + (size_t)b * NQ * ND;
    const float* crow0 = cproj + ((size_t)b * NKC + kpair * 2 + 0) * ND;
    const float* crow1 = cproj + ((size_t)b * NKC + kpair * 2 + 1) * ND;
    for (int i = tid; i < NQ * ND; i += 512) {
        const int d = i & (ND - 1);
        const float qv = qrow[i];
        As[0][i] = (_Float16)fast_sigmoid(qv + crow0[d]);
        As[1][i] = (_Float16)fast_sigmoid(qv + crow1[d]);
    }
    __syncthreads();

    const int wave = tid >> 5;       // 0..15
    const int half = wave >> 3;      // 0 or 1 -> which k
    const int wsub = wave & 7;       // n-tile group within half
    const int lane = tid & 31;
    const int low  = lane & 15;
    const int hi   = lane >> 4;

    const int k = kpair * 2 + half;
    const _Float16* Ak = &As[half][0];
    const float* gvec = g + ((size_t)b * NKC + k) * ND;

    float p0[8], p1[8], p2[8];
#pragma unroll
    for (int r = 0; r < 8; ++r) { p0[r] = 0.f; p1[r] = 0.f; p2[r] = 0.f; }

    for (int it = 0; it < 8; ++it) {
        const int ntile = it * 8 + wsub;
        const int e = ntile * 16 + low;
        const _Float16* wp = w0h + (size_t)e * ND;
        // Prefetch next iteration's W0 row into cache (global_prefetch_b8)
        if (it < 7) {
            const _Float16* wnext = w0h + (size_t)((it + 1) * 8 * 16 + wsub * 16 + low) * ND;
            __builtin_prefetch(wnext, 0, 0);
        }
        const float gc  = gvec[e];
        const float b0c = b0[e];

        v8f acc0 = {}, acc1 = {}, acc2 = {};
        const int abase = low * ND;   // row offset for this lane within each M-tile

#pragma unroll 2
        for (int kk = 0; kk < 32; ++kk) {
            const int k0 = kk * 32;
            // B frag: two 16B reads of W0 row e at k0 + hi*8 and +16 (matches A layout)
            const _Float16* bpp = wp + k0 + hi * 8;
            v16h bf = hcat(*(const v8h*)(bpp), *(const v8h*)(bpp + 16));
            // A frags from LDS: interleaved chunk pattern per 16-bit A layout
            const int ao = k0 + hi * 8;
            const _Float16* a0p = Ak + abase + ao;
            const _Float16* a1p = Ak + abase + 16 * ND + ao;
            const _Float16* a2p = Ak + abase + 32 * ND + ao;
            v16h af0 = hcat(*(const v8h*)(a0p), *(const v8h*)(a0p + 16));
            v16h af1 = hcat(*(const v8h*)(a1p), *(const v8h*)(a1p + 16));
            v16h af2 = hcat(*(const v8h*)(a2p), *(const v8h*)(a2p + 16));
            // Same B operand back-to-back: hint reuse_b on the first two
            acc0 = wmma_f16<true >(af0, bf, acc0);
            acc1 = wmma_f16<true >(af1, bf, acc1);
            acc2 = wmma_f16<false>(af2, bf, acc2);
        }
#pragma unroll
        for (int r = 0; r < 8; ++r) {
            p0[r] += fast_tanh(acc0[r] + b0c) * gc;
            p1[r] += fast_tanh(acc1[r] + b0c) * gc;
            p2[r] += fast_tanh(acc2[r] + b0c) * gc;
        }
    }

    // Reduce over the 16 column-lanes of each half-wave group
#pragma unroll
    for (int r = 0; r < 8; ++r) {
        for (int mask = 1; mask < 16; mask <<= 1) {
            p0[r] += __shfl_xor(p0[r], mask, 32);
            p1[r] += __shfl_xor(p1[r], mask, 32);
            p2[r] += __shfl_xor(p2[r], mask, 32);
        }
    }
    if (low == 0) {
        const int qoff = hi * 8;   // hi=0 -> rows r, hi=1 -> rows r+8
#pragma unroll
        for (int r = 0; r < 8; ++r) {
            scores[((size_t)b * NQ + ( 0 + qoff + r)) * NKC + k] = p0[r];
            scores[((size_t)b * NQ + (16 + qoff + r)) * NKC + k] = p1[r];
            scores[((size_t)b * NQ + (32 + qoff + r)) * NKC + k] = p2[r];
        }
    }
}

// ---------------------------------------------------------------------------
// Softmax over K + weighted context + query passthrough. One WG per (b,q).
// ---------------------------------------------------------------------------
__global__ void softmax_wcontext(const float* __restrict__ scores, const float* __restrict__ context,
                                 const float* __restrict__ query, const float* __restrict__ bcom,
                                 float* __restrict__ outQ, float* __restrict__ outWC,
                                 float* __restrict__ outA) {
    __shared__ float attn_s[NKC];
    const int bq = blockIdx.x;
    const int b  = bq / NQ;
    const int tid = threadIdx.x;

    if (tid == 0) {
        float s[NKC];
        float mx = -1e30f;
        const float bc = bcom[0];
        for (int k = 0; k < NKC; ++k) {
            s[k] = scores[(size_t)bq * NKC + k] + bc;
            mx = fmaxf(mx, s[k]);
        }
        float sum = 0.f;
        for (int k = 0; k < NKC; ++k) {
            s[k] = __builtin_amdgcn_exp2f(1.442695041f * (s[k] - mx));
            sum += s[k];
        }
        const float inv = 1.0f / sum;
        for (int k = 0; k < NKC; ++k) {
            const float a = s[k] * inv;
            attn_s[k] = a;
            outA[(size_t)bq * NKC + k] = a;
        }
    }
    __syncthreads();

    const float* ctx = context + (size_t)b * NKC * ND;
    for (int d = tid; d < ND; d += 256) {
        float acc = 0.f;
        for (int k = 0; k < NKC; ++k) acc += attn_s[k] * ctx[(size_t)k * ND + d];
        outWC[(size_t)bq * ND + d] = acc;
        outQ[(size_t)bq * ND + d]  = query[(size_t)bq * ND + d];
    }
}

// ---------------------------------------------------------------------------
extern "C" void kernel_launch(void* const* d_in, const int* in_sizes, int n_in,
                              void* d_out, int out_size, void* d_ws, size_t ws_size,
                              hipStream_t stream) {
    const float* query   = (const float*)d_in[0];
    const float* context = (const float*)d_in[1];
    const float* Wq   = (const float*)d_in[2];
    const float* bq   = (const float*)d_in[3];
    const float* Wc   = (const float*)d_in[4];
    const float* bc   = (const float*)d_in[5];
    const float* W0   = (const float*)d_in[6];
    const float* b0   = (const float*)d_in[7];
    const float* W1   = (const float*)d_in[8];
    const float* b1   = (const float*)d_in[9];
    const float* wcom = (const float*)d_in[10];
    const float* bcom = (const float*)d_in[11];

    const size_t nQD = (size_t)NB * NQ * ND;   // 1,572,864
    const size_t nKD = (size_t)NB * NKC * ND;  // 1,179,648
    const size_t nWW = (size_t)ND * ND;        // 1,048,576

    // Workspace carve-up: fp32 region, then f16 region
    float* qp = (float*)d_ws;            // qproj      (B*Q*D)
    float* cp = qp + nQD;                // cproj      (B*K*D)
    float* gg = cp + nKD;                // gate*wcom  (B*K*D)
    float* sc = gg + nKD;                // scores     (B*Q*K)
    _Float16* qh  = (_Float16*)(sc + (size_t)NB * NQ * NKC);
    _Float16* ch  = qh  + nQD;
    _Float16* wqh = ch  + nKD;
    _Float16* wch = wqh + nWW;
    _Float16* w1h = wch + nWW;
    _Float16* w0h = w1h + nWW;

    float* outQ  = (float*)d_out;
    float* outWC = outQ + nQD;
    float* outA  = outWC + nQD;

    // 1) fp32 -> f16 conversions for all GEMM operands
    convert_f16<<<(int)(nQD / 1024), 256, 0, stream>>>(query,   qh,  (int)nQD);
    convert_f16<<<(int)(nKD / 1024), 256, 0, stream>>>(context, ch,  (int)nKD);
    convert_f16<<<(int)(nWW / 1024), 256, 0, stream>>>(Wq,      wqh, (int)nWW);
    convert_f16<<<(int)(nWW / 1024), 256, 0, stream>>>(Wc,      wch, (int)nWW);
    convert_f16<<<(int)(nWW / 1024), 256, 0, stream>>>(W1,      w1h, (int)nWW);
    convert_f16<<<(int)(nWW / 1024), 256, 0, stream>>>(W0,      w0h, (int)nWW);

    // 2) Projections (WMMA f16). N-tiles = 64, 4 per block -> grid.y = 16.
    proj_gemm<<<dim3((NB * NQ) / 16, 16), 128, 0, stream>>>(qh, wqh, bq, nullptr, qp, 0);
    proj_gemm<<<dim3((NB * NKC) / 16, 16), 128, 0, stream>>>(ch, wch, bc, nullptr, cp, 0);
    proj_gemm<<<dim3((NB * NKC) / 16, 16), 128, 0, stream>>>(ch, w1h, b1, wcom,   gg, 1);

    // 3) Main fused GEMM + score reduction: one WG per (b, k-pair), 16 waves
    fused_score_gemm<<<NB * (NKC / 2), 512, 0, stream>>>(qp, cp, gg, w0h, b0, sc);

    // 4) Softmax over K + weighted context + query copy: one WG per (b,q)
    softmax_wcontext<<<NB * NQ, 256, 0, stream>>>(sc, context, query, bcom, outQ, outWC, outA);
}
